// MultiHeadAttention_58540404244495
// MI455X (gfx1250) — matrix operand order repaired and checked by reading, not compiled, not run on hardware
//
#include <hip/hip_runtime.h>
#include <hip/hip_bf16.h>

#define B_  2
#define T_  2048
#define C_  1024
#define H_  16
#define HD_ 64
#define M_  (B_ * T_)   // 4096

typedef __attribute__((ext_vector_type(16))) __bf16 v16bf;
typedef __attribute__((ext_vector_type(8)))  __bf16 v8bf;
typedef __attribute__((ext_vector_type(4)))  __bf16 v4bf;
typedef __attribute__((ext_vector_type(8)))  float  v8f;
typedef __attribute__((ext_vector_type(4)))  unsigned int u32x4;
typedef __attribute__((ext_vector_type(8)))  int i32x8;
typedef __attribute__((ext_vector_type(4)))  int i32x4;

__device__ __forceinline__ v16bf cat16(v8bf lo, v8bf hi) {
    v16bf r;
#pragma unroll
    for (int i = 0; i < 8; ++i) { r[i] = lo[i]; r[i + 8] = hi[i]; }
    return r;
}

__device__ __forceinline__ v8f wmma_bf16(v16bf a, v16bf b, v8f c) {
    return __builtin_amdgcn_wmma_f32_16x16x32_bf16(false, a, false, b, (short)0, c, false, false);
}

// ---------------------------------------------------------------- TDM tile load
// Builds a D# (groups per CDNA5 ISA) for a 2D bf16 tile
// (tile1 rows x tile0 elems, global row stride stride0 elems) -> LDS with
// per-row padding (pad codes: interval 2^(pi+1) DWORDs, amount pa+1 DWORDs).
// tile1==0 -> 1D tile of tile0 elems. Issue once per workgroup (wave 0).
// This toolchain exposes the 6-arg builtin: (g0, g1, g2, g3, g4, cpol).
__device__ __forceinline__ void tdm_load_bf16(unsigned lds_addr, const void* gaddr,
                                              unsigned tile0, unsigned tile1,
                                              unsigned tdim0, unsigned tdim1,
                                              unsigned long long stride0,
                                              unsigned pi, unsigned pa) {
    unsigned long long ga = (unsigned long long)(size_t)gaddr;
    u32x4 g0;
    g0[0] = 1u;                                                  // count=1, user D#
    g0[1] = lds_addr;                                            // LDS byte address
    g0[2] = (unsigned)ga;                                        // global addr [31:0]
    g0[3] = (unsigned)((ga >> 32) & 0x01FFFFFFu) | (2u << 30);   // [56:32] | type=2
    i32x8 g1;
    g1[0] = (int)((1u << 16)            // data_size = 2 bytes
                | (1u << 20)            // pad_enable
                | (pi << 22) | (pa << 25));
    g1[1] = (int)((tdim0 & 0xFFFFu) << 16);                          // tensor_dim0[15:0]
    g1[2] = (int)(((tdim0 >> 16) & 0xFFFFu) | ((tdim1 & 0xFFFFu) << 16));
    g1[3] = (int)(((tdim1 >> 16) & 0xFFFFu) | (tile0 << 16));        // tile_dim0
    g1[4] = (int)(tile1 & 0xFFFFu);                                  // tile_dim1 (tile_dim2=0)
    g1[5] = (int)(unsigned)(stride0 & 0xFFFFFFFFu);                  // dim0 stride [31:0]
    g1[6] = (int)(unsigned)((stride0 >> 32) & 0xFFFFu);              // [47:32] (dim1 stride=0)
    g1[7] = 0;
    i32x4 z4 = {0, 0, 0, 0};
    i32x8 z8 = {0, 0, 0, 0, 0, 0, 0, 0};
    __builtin_amdgcn_tensor_load_to_lds(g0, g1, z4, z4, z8, 0);
}

// ---------------------------------------------------------------- cvt f32->bf16
__global__ void cvt4(const float4* __restrict__ s, __bf16* __restrict__ d, int n4) {
    int i = blockIdx.x * 256 + threadIdx.x;
    if (i < n4) {
        float4 v = s[i];
        v4bf o;
        o[0] = (__bf16)v.x; o[1] = (__bf16)v.y; o[2] = (__bf16)v.z; o[3] = (__bf16)v.w;
        *(v4bf*)(d + (size_t)i * 4) = o;
    }
}

// ---------------------------------------------------------------- WMMA GEMM
// C[M x NW] = A[M x 1024] * W[1024 x NW] + bias ; per-WG tile 128x64, K-step 32.
// A tile staged by TDM; B tile staged transposed by the waves (overlaps TDM).
// mode 0: scatter to Q/K/V bf16 [B,H,T,64] ; mode 1: f32 out [M x 1024].
__launch_bounds__(256)
__global__ void gemm_wmma(const __bf16* __restrict__ A, const __bf16* __restrict__ W,
                          const float* __restrict__ bias, const int NW, const int mode,
                          float* __restrict__ outf,
                          __bf16* __restrict__ qb, __bf16* __restrict__ kb,
                          __bf16* __restrict__ vb) {
    __shared__ __align__(16) __bf16 As[128][40];   // [m][k], TDM-padded 32->40
    __shared__ __align__(16) __bf16 Bs[64][40];    // [n][k] (transposed), padded

    const int tid  = threadIdx.x;
    const int lane = tid & 31;
    const int wave = tid >> 5;
    const int wm   = wave & 3;          // 4 waves along M
    const int wn   = wave >> 2;         // 2 waves along N
    const int m0   = blockIdx.y * 128;
    const int n0   = blockIdx.x * 64;
    const int hl   = lane >> 4;         // half-wave select
    const int l16  = lane & 15;

    v8f acc[2][2];
    v8f zf = {};
#pragma unroll
    for (int i = 0; i < 2; ++i)
#pragma unroll
        for (int j = 0; j < 2; ++j) acc[i][j] = zf;

    const unsigned as_lds = (unsigned)(size_t)(void*)&As[0][0];

    for (int k0 = 0; k0 < C_; k0 += 32) {
        __syncthreads();
        // ---- A tile (128x32, row stride C_) via Tensor Data Mover
        if (wave == 0)
            tdm_load_bf16(as_lds, A + (size_t)m0 * C_ + k0,
                          /*tile0*/32, /*tile1*/128, /*tdim0*/32, /*tdim1*/128,
                          /*stride0*/(unsigned long long)C_,
                          /*pi: 16 DW*/3, /*pa: 4 DW*/3);
        // ---- B tile (32x64) transposed into [n][k] by all waves
        {
            int kr = tid >> 3;   // 0..31
            int nv = tid & 7;    // 0..7
            v8bf v = *(const v8bf*)(W + (size_t)(k0 + kr) * NW + n0 + nv * 8);
#pragma unroll
            for (int i = 0; i < 8; ++i) Bs[nv * 8 + i][kr] = v[i];
        }
        if (k0 + 32 < C_)
            __builtin_prefetch(W + (size_t)(k0 + 32 + (tid >> 3)) * NW + n0, 0, 0);
        if (wave == 0) __builtin_amdgcn_s_wait_tensorcnt(0);
        __syncthreads();

        // ---- fragments (ISA wave32 layouts) + 4 WMMAs
        v16bf af[2], bfr[2];
#pragma unroll
        for (int f = 0; f < 2; ++f) {
            int row = wm * 32 + f * 16 + l16;
            int klo = hl ? 8 : 0;
            v8bf lo = *(const v8bf*)&As[row][klo];
            v8bf hi = *(const v8bf*)&As[row][klo + 16];
            af[f] = cat16(lo, hi);
        }
#pragma unroll
        for (int f = 0; f < 2; ++f) {
            int ncol = wn * 32 + f * 16 + l16;
            int klo  = hl ? 8 : 0;
            v8bf lo = *(const v8bf*)&Bs[ncol][klo];
            v8bf hi = *(const v8bf*)&Bs[ncol][klo + 16];
            bfr[f] = cat16(lo, hi);
        }
#pragma unroll
        for (int i = 0; i < 2; ++i)
#pragma unroll
            for (int j = 0; j < 2; ++j)
                acc[i][j] = wmma_bf16(af[i], bfr[j], acc[i][j]);
    }

    // ---- epilogue: D layout m = v + 8*hl (per VGPR v), n = lane%16
#pragma unroll
    for (int i = 0; i < 2; ++i) {
#pragma unroll
        for (int j = 0; j < 2; ++j) {
            int n_g  = n0 + wn * 32 + j * 16 + l16;
            float bv = bias[n_g];
#pragma unroll
            for (int vv = 0; vv < 8; ++vv) {
                int m_g   = m0 + wm * 32 + i * 16 + vv + hl * 8;
                float val = acc[i][j][vv] + bv;
                if (mode == 0) {
                    int which = n_g >> 10;
                    int nn    = n_g & 1023;
                    int h     = nn >> 6, d = nn & 63;
                    int bb    = m_g >> 11, t = m_g & 2047;
                    size_t off = (((size_t)bb * H_ + h) * T_ + t) * HD_ + d;
                    __bf16* dst = (which == 0) ? qb : ((which == 1) ? kb : vb);
                    dst[off] = (__bf16)val;
                } else {
                    outf[(size_t)m_g * C_ + n_g] = val;
                }
            }
        }
    }
}

// ---------------------------------------------------------------- flash attention
// One WG per (b*h, 128 q rows); wave w owns q rows [q0+16w, q0+16w+16).
// K tile staged by TDM (32 contiguous rows of 64 bf16, padded 64->72);
// V tile staged transposed by the waves (overlaps the TDM transfer).
__launch_bounds__(256)
__global__ void flash_attn(const __bf16* __restrict__ Q, const __bf16* __restrict__ K,
                           const __bf16* __restrict__ V, __bf16* __restrict__ O) {
    __shared__ __align__(16) __bf16 Ks[32][72];      // [key][d], TDM-padded
    __shared__ __align__(16) __bf16 Vs[64][40];      // [d][key] (transposed)
    __shared__ __align__(16) __bf16 Ps[8][16][40];   // per-wave P staging [m][k]

    const int tid  = threadIdx.x, lane = tid & 31, wave = tid >> 5;
    const int hl   = lane >> 4, l16 = lane & 15;
    const int bh   = blockIdx.y;
    const int bb   = bh >> 4, h = bh & 15;
    const int q0   = blockIdx.x * 128;
    const __bf16* qp = Q + (size_t)bh * T_ * HD_;
    const __bf16* kp = K + (size_t)bh * T_ * HD_;
    const __bf16* vp = V + (size_t)bh * T_ * HD_;
    const unsigned ks_lds = (unsigned)(size_t)(void*)&Ks[0][0];

    // Q rows as two 16x32 A fragments, kept in registers all kernel
    v16bf aq[2];
#pragma unroll
    for (int f = 0; f < 2; ++f) {
        int t   = q0 + wave * 16 + l16;
        int dlo = f * 32 + (hl ? 8 : 0);
        v8bf lo = *(const v8bf*)(qp + (size_t)t * HD_ + dlo);
        v8bf hi = *(const v8bf*)(qp + (size_t)t * HD_ + dlo + 16);
        aq[f] = cat16(lo, hi);
    }

    v8f o[4];
    v8f zf = {};
#pragma unroll
    for (int i = 0; i < 4; ++i) o[i] = zf;
    float m_i[8], l_i[8];
#pragma unroll
    for (int i = 0; i < 8; ++i) { m_i[i] = -3.0e38f; l_i[i] = 0.f; }

    const int nj = (q0 + 128) / 32;       // uniform across WG -> barriers legal
    for (int j = 0; j < nj; ++j) {
        __syncthreads();
        // ---- K tile via TDM: one contiguous 4KB block, padded into [32][72]
        if (wave == 0)
            tdm_load_bf16(ks_lds, kp + (size_t)j * 32 * HD_,
                          /*tile0*/32 * HD_, /*tile1*/0,
                          /*tdim0*/32 * HD_, /*tdim1*/0,
                          /*stride0*/0ull, /*pi: 32 DW*/4, /*pa: 4 DW*/3);
        // ---- V tile transposed into [d][key] by all waves
        {
            int key = tid >> 3, dv = tid & 7;
            v8bf vv = *(const v8bf*)(vp + (size_t)(j * 32 + key) * HD_ + dv * 8);
#pragma unroll
            for (int i = 0; i < 8; ++i) Vs[dv * 8 + i][key] = vv[i];
        }
        if (wave == 0) __builtin_amdgcn_s_wait_tensorcnt(0);
        __syncthreads();

        // S = Q K^T : 16q x 32keys, contraction over d=64 (2 WMMA chains x 2 frags)
        v8f s[2];
#pragma unroll
        for (int fn = 0; fn < 2; ++fn) {
            s[fn] = zf;
#pragma unroll
            for (int kk = 0; kk < 2; ++kk) {
                int keyl = fn * 16 + l16;
                int dlo  = kk * 32 + (hl ? 8 : 0);
                v8bf lo = *(const v8bf*)&Ks[keyl][dlo];
                v8bf hi = *(const v8bf*)&Ks[keyl][dlo + 16];
                s[fn] = wmma_bf16(aq[kk], cat16(lo, hi), s[fn]);
            }
        }

        // online softmax (rows replicated across 16-lane halves)
        float p0[8], p1[8];
#pragma unroll
        for (int vv = 0; vv < 8; ++vv) {
            int qg  = q0 + wave * 16 + vv + hl * 8;
            int kg0 = j * 32 + l16;
            int kg1 = j * 32 + 16 + l16;
            float s0 = s[0][vv] * 0.125f;
            float s1 = s[1][vv] * 0.125f;
            bool msk0 = kg0 > qg, msk1 = kg1 > qg;
            s0 = msk0 ? -3.0e38f : s0;
            s1 = msk1 ? -3.0e38f : s1;
            float rm = fmaxf(s0, s1);
#pragma unroll
            for (int m = 1; m < 16; m <<= 1) rm = fmaxf(rm, __shfl_xor(rm, m, 32));
            float mn = fmaxf(m_i[vv], rm);
            float e0 = msk0 ? 0.f : __expf(s0 - mn);
            float e1 = msk1 ? 0.f : __expf(s1 - mn);
            float rs = e0 + e1;
#pragma unroll
            for (int m = 1; m < 16; m <<= 1) rs += __shfl_xor(rs, m, 32);
            float alpha = __expf(m_i[vv] - mn);
            l_i[vv] = l_i[vv] * alpha + rs;
            m_i[vv] = mn;
#pragma unroll
            for (int fo = 0; fo < 4; ++fo) o[fo][vv] *= alpha;
            p0[vv] = e0; p1[vv] = e1;
        }

        // reshuffle P: D layout -> A layout through per-wave LDS staging
#pragma unroll
        for (int vv = 0; vv < 8; ++vv) {
            int r = vv + hl * 8;
            Ps[wave][r][l16]      = (__bf16)p0[vv];
            Ps[wave][r][16 + l16] = (__bf16)p1[vv];
        }
        __syncthreads();
        v16bf ap;
        {
            int klo = hl ? 8 : 0;
            v8bf lo = *(const v8bf*)&Ps[wave][l16][klo];
            v8bf hi = *(const v8bf*)&Ps[wave][l16][klo + 16];
            ap = cat16(lo, hi);
        }

        // O += P V : contraction over 32 keys, 4 d-fragments
#pragma unroll
        for (int fo = 0; fo < 4; ++fo) {
            int dl  = fo * 16 + l16;
            int klo = hl ? 8 : 0;
            v8bf lo = *(const v8bf*)&Vs[dl][klo];
            v8bf hi = *(const v8bf*)&Vs[dl][klo + 16];
            o[fo] = wmma_bf16(ap, cat16(lo, hi), o[fo]);
        }
    }

    // epilogue: O / l -> attn buffer [B,T,C] with head columns h*64+d
#pragma unroll
    for (int fo = 0; fo < 4; ++fo) {
#pragma unroll
        for (int vv = 0; vv < 8; ++vv) {
            int t   = q0 + wave * 16 + vv + hl * 8;
            int col = h * HD_ + fo * 16 + l16;
            O[((size_t)bb * T_ + t) * C_ + col] = (__bf16)(o[fo][vv] / l_i[vv]);
        }
    }
}

// ---------------------------------------------------------------- launch
extern "C" void kernel_launch(void* const* d_in, const int* in_sizes, int n_in,
                              void* d_out, int out_size, void* d_ws, size_t ws_size,
                              hipStream_t stream) {
    const float* x      = (const float*)d_in[0];
    const float* w_qkv  = (const float*)d_in[1];
    const float* b_qkv  = (const float*)d_in[2];
    const float* w_proj = (const float*)d_in[3];
    const float* b_proj = (const float*)d_in[4];
    float* out = (float*)d_out;

    const size_t nx   = (size_t)M_ * C_;            // 4 Mi
    const size_t nwq  = (size_t)C_ * 3 * C_;        // 3 Mi
    const size_t nwp  = (size_t)C_ * C_;            // 1 Mi
    const size_t nqkv = (size_t)B_ * H_ * T_ * HD_; // 4 Mi
    __bf16* ws  = (__bf16*)d_ws;
    __bf16* xb  = ws;
    __bf16* wqb = xb + nx;
    __bf16* wpb = wqb + nwq;
    __bf16* Qb  = wpb + nwp;
    __bf16* Kb  = Qb + nqkv;
    __bf16* Vb  = Kb + nqkv;
    __bf16* Ab  = Vb + nqkv;   // ~48 MB total bf16 workspace

    cvt4<<<(int)(nx  / 4 + 255) / 256, 256, 0, stream>>>((const float4*)x,      xb,  (int)(nx  / 4));
    cvt4<<<(int)(nwq / 4 + 255) / 256, 256, 0, stream>>>((const float4*)w_qkv,  wqb, (int)(nwq / 4));
    cvt4<<<(int)(nwp / 4 + 255) / 256, 256, 0, stream>>>((const float4*)w_proj, wpb, (int)(nwp / 4));

    gemm_wmma<<<dim3(3 * C_ / 64, M_ / 128), 256, 0, stream>>>(
        xb, wqb, b_qkv, 3 * C_, 0, nullptr, Qb, Kb, Vb);

    flash_attn<<<dim3(T_ / 128, B_ * H_), 256, 0, stream>>>(Qb, Kb, Vb, Ab);

    gemm_wmma<<<dim3(C_ / 64, M_ / 128), 256, 0, stream>>>(
        Ab, wpb, b_proj, C_, 1, out, nullptr, nullptr, nullptr);
}
